// SimpleGNN_22591527977361
// MI455X (gfx1250) — matrix operand order repaired and checked by reading, not compiled, not run on hardware
//
#include <hip/hip_runtime.h>
#include <hip/hip_bf16.h>
#include <math.h>

typedef __attribute__((ext_vector_type(16))) _Float16 v16h;
typedef __attribute__((ext_vector_type(8)))  float    v8f;

#define HID 132          /* padded LDS row stride (halves): 264B -> bank stride 2, conflict-free */
#define MTILE 64         /* nodes per block iteration */

__device__ inline v16h z16() {
  v16h r;
#pragma unroll
  for (int i = 0; i < 16; ++i) r[i] = (_Float16)0.0f;
  return r;
}
__device__ inline v8f z8() {
  v8f r;
#pragma unroll
  for (int i = 0; i < 8; ++i) r[i] = 0.0f;
  return r;
}

// B fragment (32x16 f16) for output column n, K window [k0, k0+32).
// Lane layout: lanes 0-15 hold K=k0..k0+15, lanes 16-31 hold K=k0+16..k0+31 (sequential).
__device__ inline v16h load_bfrag(const float* __restrict__ W, int n, int ghalf, int k0) {
  v16h b;
  const int kb = k0 + ghalf * 16;
#pragma unroll
  for (int i = 0; i < 16; ++i) b[i] = (_Float16)W[(kb + i) * 128 + n];
  return b;
}

// A fragment (16x32 f16) from an LDS row pointer (already offset to k-window start).
// Lane-half g: VGPR v -> K = g*16 + 2v (sequential, matches B fragment).
__device__ inline v16h load_afrag_lds(const _Float16* __restrict__ rowp, int ghalf) {
  v16h a;
#pragma unroll
  for (int v = 0; v < 8; ++v) {
    const int kb = ghalf * 16 + (v << 1);
    a[2 * v]     = rowp[kb];
    a[2 * v + 1] = rowp[kb + 1];
  }
  return a;
}

// ---------------- kernel 1: zero workspace ----------------
__global__ void gnn_zero(float* __restrict__ p, long n) {
  long i = (long)blockIdx.x * blockDim.x + threadIdx.x;
  long stride = (long)gridDim.x * blockDim.x;
  for (; i < n; i += stride) p[i] = 0.0f;
}

// ---------------- kernel 2: edge scatter (deg + agg) ----------------
__global__ void gnn_scatter(const int* __restrict__ ei, const float* __restrict__ x,
                            float* __restrict__ agg, float* __restrict__ deg, int E) {
  long e = (long)blockIdx.x * blockDim.x + threadIdx.x;
  long stride = (long)gridDim.x * blockDim.x;
  const float4* __restrict__ x4 = (const float4*)x;
  for (; e < E; e += stride) {
    const int s = ei[e];
    const int d = ei[(long)E + e];
    atomicAdd(deg + s, 1.0f);
    const float4 lo = x4[(long)d * 2];
    const float4 hi = x4[(long)d * 2 + 1];
    float* ap = agg + (long)s * 8;
    atomicAdd(ap + 0, lo.x); atomicAdd(ap + 1, lo.y);
    atomicAdd(ap + 2, lo.z); atomicAdd(ap + 3, lo.w);
    atomicAdd(ap + 4, hi.x); atomicAdd(ap + 5, hi.y);
    atomicAdd(ap + 6, hi.z); atomicAdd(ap + 7, hi.w);
  }
}

// ---------------- kernel 3: fused 3-layer MLP via WMMA ----------------
__global__ void __launch_bounds__(256, 2)
gnn_mlp(const float* __restrict__ x, const float* __restrict__ agg,
        const float* __restrict__ deg,
        const float* __restrict__ W1, const float* __restrict__ b1,
        const float* __restrict__ W2, const float* __restrict__ b2,
        const float* __restrict__ W3, const float* __restrict__ b3,
        float* __restrict__ colsum, int nNodes) {
  __shared__ _Float16 sHa[MTILE * HID];
  __shared__ _Float16 sHb[MTILE * HID];

  const int lane  = threadIdx.x & 31;
  const int wave  = threadIdx.x >> 5;
  const int nloc  = lane & 15;
  const int ghalf = lane >> 4;
  const int n     = wave * 16 + nloc;   // output column owned by this lane

  // ---- resident B fragments (built once per block, weights are L2-resident) ----
  v16h bf1 = z16();
  if (ghalf == 0) {
#pragma unroll
    for (int i = 0; i < 8; ++i) bf1[i] = (_Float16)W1[i * 128 + n];  // K=8 zero-padded to 32
  }
  v16h bf2[4], bf3[4];
#pragma unroll
  for (int ks = 0; ks < 4; ++ks) {
    bf2[ks] = load_bfrag(W2, n, ghalf, ks * 32);
    bf3[ks] = load_bfrag(W3, n, ghalf, ks * 32);
  }
  const float bias1 = b1[n], bias2 = b2[n], bias3 = b3[n];

  float acc = 0.0f;  // masked column sum of h3 for this lane's column

  const int tileStride = gridDim.x * MTILE;
  for (int base = blockIdx.x * MTILE; base < nNodes; base += tileStride) {
    // ---- layer 1: Z0 = x + agg/deg (K=8, padded) -> H1 in sHa ----
#pragma unroll
    for (int mt = 0; mt < 4; ++mt) {
      v16h a = z16();
      const int node = base + mt * 16 + (lane & 15);
      if (ghalf == 0 && node < nNodes) {
        const float* xp = x   + (long)node * 8;
        const float* ap = agg + (long)node * 8;
        const float invd = 1.0f / fmaxf(deg[node], 1.0f);
#pragma unroll
        for (int i = 0; i < 8; ++i) a[i] = (_Float16)(xp[i] + ap[i] * invd);
      }
      v8f c = __builtin_amdgcn_wmma_f32_16x16x32_f16(false, a, false, bf1,
                                                     (short)0, z8(), false, false);
#pragma unroll
      for (int r = 0; r < 8; ++r) {
        const float hv = fmaxf(c[r] + bias1, 0.0f);
        sHa[(mt * 16 + r + 8 * ghalf) * HID + n] = (_Float16)hv;
      }
    }
    __syncthreads();

    // ---- layer 2: H1 @ W2 -> H2 in sHb ----
#pragma unroll
    for (int mt = 0; mt < 4; ++mt) {
      v8f c = z8();
      const _Float16* rowp = sHa + (mt * 16 + (lane & 15)) * HID;
#pragma unroll
      for (int ks = 0; ks < 4; ++ks) {
        v16h a = load_afrag_lds(rowp + ks * 32, ghalf);
        c = __builtin_amdgcn_wmma_f32_16x16x32_f16(false, a, false, bf2[ks],
                                                   (short)0, c, false, false);
      }
#pragma unroll
      for (int r = 0; r < 8; ++r) {
        const float hv = fmaxf(c[r] + bias2, 0.0f);
        sHb[(mt * 16 + r + 8 * ghalf) * HID + n] = (_Float16)hv;
      }
    }
    __syncthreads();

    // ---- layer 3: H2 @ W3 -> masked column-sum accumulation ----
#pragma unroll
    for (int mt = 0; mt < 4; ++mt) {
      v8f c = z8();
      const _Float16* rowp = sHb + (mt * 16 + (lane & 15)) * HID;
#pragma unroll
      for (int ks = 0; ks < 4; ++ks) {
        v16h a = load_afrag_lds(rowp + ks * 32, ghalf);
        c = __builtin_amdgcn_wmma_f32_16x16x32_f16(false, a, false, bf3[ks],
                                                   (short)0, c, false, false);
      }
#pragma unroll
      for (int r = 0; r < 8; ++r) {
        const int node = base + mt * 16 + r + 8 * ghalf;
        if (node < nNodes) acc += fmaxf(c[r] + bias3, 0.0f);
      }
    }
    // loop-around barrier: protects sHb readers vs next tile's writers,
    // and sHa writers (next L1) run disjoint from completed L2 reads.
  }

  atomicAdd(colsum + n, acc);
}

// ---------------- kernel 4: mean + head ----------------
__global__ void gnn_head(const float* __restrict__ colsum, const float* __restrict__ Wv,
                         const float* __restrict__ bv, float* __restrict__ out, int nNodes) {
  const int lane = threadIdx.x;  // 32 threads
  float s = 0.0f;
#pragma unroll
  for (int k = 0; k < 4; ++k) {
    const int nn = lane + 32 * k;
    s += colsum[nn] * Wv[nn];
  }
#pragma unroll
  for (int off = 16; off > 0; off >>= 1) s += __shfl_xor(s, off, 32);
  if (lane == 0) out[0] = tanhf(s / (float)nNodes + bv[0]);
}

extern "C" void kernel_launch(void* const* d_in, const int* in_sizes, int n_in,
                              void* d_out, int out_size, void* d_ws, size_t ws_size,
                              hipStream_t stream) {
  const float* x  = (const float*)d_in[0];
  const int*   ei = (const int*)d_in[1];
  const float* W1 = (const float*)d_in[2];
  const float* b1 = (const float*)d_in[3];
  const float* W2 = (const float*)d_in[4];
  const float* b2 = (const float*)d_in[5];
  const float* W3 = (const float*)d_in[6];
  const float* b3 = (const float*)d_in[7];
  const float* Wv = (const float*)d_in[8];
  const float* bv = (const float*)d_in[9];

  const int N = in_sizes[0] / 8;   // nodes
  const int E = in_sizes[1] / 2;   // edges

  float* agg    = (float*)d_ws;            // N*8
  float* deg    = agg + (long)N * 8;       // N
  float* colsum = deg + N;                 // 128
  const long zn = (long)N * 8 + N + 128;

  gnn_zero<<<1024, 256, 0, stream>>>(agg, zn);
  gnn_scatter<<<4096, 256, 0, stream>>>(ei, x, agg, deg, E);

  const int tiles = (N + MTILE - 1) / MTILE;
  const int grid  = tiles < 1920 ? tiles : 1920;
  gnn_mlp<<<grid, 256, 0, stream>>>(x, agg, deg, W1, b1, W2, b2, W3, b3, colsum, N);

  gnn_head<<<1, 32, 0, stream>>>(colsum, Wv, bv, (float*)d_out, N);
}
